// MHSelfAttention_57389353009639
// MI455X (gfx1250) — matrix operand order repaired and checked by reading, not compiled, not run on hardware
//
#include <hip/hip_runtime.h>
#include <hip/hip_bf16.h>

// MI455X / gfx1250 MHSA: bf16 WMMA GEMMs + flash-attention streaming softmax.
// Matrix math on v_wmma_f32_16x16x32_bf16 (wave32). Tile staging uses
// GLOBAL_LOAD_ASYNC_TO_LDS_B128 (ASYNCcnt) with double-buffered LDS when the
// toolchain exposes the builtin; sync fallback otherwise.

typedef __bf16 bf16_t;
typedef __attribute__((ext_vector_type(16))) __bf16 v16bf;
typedef __attribute__((ext_vector_type(8)))  __bf16 v8bf;
typedef __attribute__((ext_vector_type(8)))  float  v8f;
typedef __attribute__((ext_vector_type(4)))  int    v4i;
typedef __attribute__((ext_vector_type(4)))  float  f4;   // 16-byte copy unit

#define SEQ    4096
#define DMODEL 1024
#define NHEAD  16
#define DHEAD  64

#if defined(__has_builtin)
#if __has_builtin(__builtin_amdgcn_global_load_async_to_lds_b128)
#define HAVE_ASYNC_LDS 1
#endif
#endif

// Builtin signature (from hipcc diagnostic): (AS1 v4i*, AS3 v4i*, imm, imm)
typedef __attribute__((address_space(1))) v4i as1_v4i;
typedef __attribute__((address_space(3))) v4i as3_v4i;

// 16B global -> LDS copy. Async (no VGPR round trip, ASYNCcnt) when available.
__device__ __forceinline__ void cp_async16(bf16_t* lds_dst, const bf16_t* gsrc) {
#ifdef HAVE_ASYNC_LDS
  __builtin_amdgcn_global_load_async_to_lds_b128(
      (as1_v4i*)(unsigned long long)gsrc,
      (as3_v4i*)(unsigned int)(unsigned long long)lds_dst, 0, 0);
#else
  *(f4*)lds_dst = *(const f4*)gsrc;
#endif
}

__device__ __forceinline__ void wait_async_all() {
#ifdef HAVE_ASYNC_LDS
#if __has_builtin(__builtin_amdgcn_s_wait_asynccnt)
  __builtin_amdgcn_s_wait_asynccnt(0);
#else
  asm volatile("s_wait_asynccnt 0" ::: "memory");
#endif
#endif
}

__device__ __forceinline__ v16bf pack16(v8bf lo, v8bf hi) {
  return __builtin_shufflevector(lo, hi, 0,1,2,3,4,5,6,7,8,9,10,11,12,13,14,15);
}

__device__ __forceinline__ v8f wmma_bf16(v16bf a, v16bf b, v8f c) {
  // 8 args: (neg_a, A, neg_b, B, c_mod, C, reuse_a, reuse_b)
  return __builtin_amdgcn_wmma_f32_16x16x32_bf16(false, a, false, b, (short)0, c, false, false);
}

// A fragment (16x32, MxK): lane m = lane%16 holds row m; elements are two
// contiguous 8-wide K chunks at koff + {h*8, 16+h*8}, h = lane/16.
__device__ __forceinline__ v16bf load_a_frag(const bf16_t* base, int ldm, int lane, int koff) {
  const int m = lane & 15, h = lane >> 4;
  const bf16_t* r = base + (size_t)m * ldm + koff;
  v8bf lo = *(const v8bf*)(r + h * 8);
  v8bf hi = *(const v8bf*)(r + 16 + h * 8);
  return pack16(lo, hi);
}

// B fragment (32x16, KxN) supplied as its transpose (row n = B[.][n] over k):
// lane n = lane%16 holds column n; elements are two 8-wide K chunks at koff + {h*8, 16+h*8}.
__device__ __forceinline__ v16bf load_bt_frag(const bf16_t* base, int ldn, int lane, int koff) {
  const int n = lane & 15, h = lane >> 4;
  const bf16_t* r = base + (size_t)n * ldn + koff;
  v8bf lo = *(const v8bf*)(r + h * 8);
  v8bf hi = *(const v8bf*)(r + 16 + h * 8);
  return pack16(lo, hi);
}

__device__ __forceinline__ float redmax16(float v) {
  v = fmaxf(v, __shfl_xor(v, 1, 16));
  v = fmaxf(v, __shfl_xor(v, 2, 16));
  v = fmaxf(v, __shfl_xor(v, 4, 16));
  v = fmaxf(v, __shfl_xor(v, 8, 16));
  return v;
}
__device__ __forceinline__ float redsum16(float v) {
  v += __shfl_xor(v, 1, 16);
  v += __shfl_xor(v, 2, 16);
  v += __shfl_xor(v, 4, 16);
  v += __shfl_xor(v, 8, 16);
  return v;
}

// ---------------------------------------------------------------- convert ---
__global__ __launch_bounds__(256) void cvt_f32_bf16(const float* __restrict__ in,
                                                    bf16_t* __restrict__ out, int n) {
  for (int i = blockIdx.x * 256 + threadIdx.x; i < n; i += gridDim.x * 256)
    out[i] = (bf16_t)in[i];
}

// ------------------------------------------------------------------- GEMM ---
// y = A(MxK) * W^T, W stored [N][K] row-major. M=4096, N=K=1024.
// Block = 8 waves = 256(M) x 64(N); per wave 32x64 = 8 accum frags, so 8 WMMAs
// reuse 4 B-frags per 32-k step. Double-buffered async LDS staging.
// mode 0: Q -> bf16 flat, *0.125   mode 1: K -> bf16 flat
// mode 2: V -> bf16 transposed [1024][4096]   mode 3: O -> f32 flat (d_out)
__global__ __launch_bounds__(256) void gemm_bf16(const bf16_t* __restrict__ A,
                                                 const bf16_t* __restrict__ W,
                                                 bf16_t* __restrict__ outb,
                                                 float* __restrict__ outf,
                                                 int mode) {
  __shared__ __align__(16) bf16_t As[2][256][40];  // 2 x 20 KB
  __shared__ __align__(16) bf16_t Bs[2][64][40];   // 2 x  5 KB

  const int t = threadIdx.x, lane = t & 31, wave = t >> 5;
  const int mb = blockIdx.y, nb = blockIdx.x;

  v8f c[2][4];
  #pragma unroll
  for (int g = 0; g < 2; ++g)
    #pragma unroll
    for (int j = 0; j < 4; ++j) c[g][j] = (v8f)(0.0f);

  auto stage = [&](int buf, int k0) {
    #pragma unroll
    for (int i = 0; i < 4; ++i) {              // A: 256 rows x 32 k = 1024 chunks
      int ch = t + i * 256;
      int r = ch >> 2, o = (ch & 3) * 8;
      cp_async16(&As[buf][r][o], &A[(size_t)(mb * 256 + r) * DMODEL + k0 + o]);
    }
    int r = t >> 2, o = (t & 3) * 8;           // B: 64 rows x 32 k = 256 chunks
    cp_async16(&Bs[buf][r][o], &W[(size_t)(nb * 64 + r) * DMODEL + k0 + o]);
  };

  stage(0, 0);
  const int NSTEP = DMODEL / 32;
  for (int s = 0; s < NSTEP; ++s) {
    const int cur = s & 1;
    wait_async_all();        // own async stores to LDS for tile s complete
    __syncthreads();         // everyone's tile-s data visible; prev reads done
    if (s + 1 < NSTEP) stage(cur ^ 1, (s + 1) * 32);   // overlap with compute

    v16bf a0 = load_a_frag(&As[cur][wave * 32][0],      40, lane, 0);
    v16bf a1 = load_a_frag(&As[cur][wave * 32 + 16][0], 40, lane, 0);
    #pragma unroll
    for (int j = 0; j < 4; ++j) {
      v16bf b = load_bt_frag(&Bs[cur][j * 16][0], 40, lane, 0);
      c[0][j] = wmma_bf16(a0, b, c[0][j]);
      c[1][j] = wmma_bf16(a1, b, c[1][j]);
    }
  }

  // C layout: vgpr r, lane l -> row = r + 8*(l/16), col = l%16
  const int nn = lane & 15, half = lane >> 4;
  #pragma unroll
  for (int g = 0; g < 2; ++g) {
    const int m0 = mb * 256 + wave * 32 + g * 16;
    #pragma unroll
    for (int j = 0; j < 4; ++j) {
      const int col = nb * 64 + j * 16 + nn;
      #pragma unroll
      for (int r = 0; r < 8; ++r) {
        const int row = m0 + r + 8 * half;
        float v = c[g][j][r];
        if (mode == 0)      outb[(size_t)row * DMODEL + col] = (bf16_t)(v * 0.125f);
        else if (mode == 1) outb[(size_t)row * DMODEL + col] = (bf16_t)v;
        else if (mode == 2) outb[(size_t)col * SEQ + row]    = (bf16_t)v;
        else                outf[(size_t)row * DMODEL + col] = v;
      }
    }
  }
}

// -------------------------------------------------------- flash attention ---
// Block = (head h, 128 q rows). 8 waves, each owns 16 q rows x dh=64.
// Streams 32-key tiles of K and V^T through double-buffered async LDS.
__global__ __launch_bounds__(256) void attn_flash(const bf16_t* __restrict__ Qb,
                                                  const bf16_t* __restrict__ Kb,
                                                  const bf16_t* __restrict__ Vtb,
                                                  bf16_t* __restrict__ attnb) {
  __shared__ __align__(16) bf16_t Ks[2][32][72];      // 32 keys x 64 d (+pad)
  __shared__ __align__(16) bf16_t Vts[2][64][40];     // 64 d x 32 keys (+pad)
  __shared__ __align__(16) bf16_t Plds[8][16][40];    // per-wave P bounce

  const int t = threadIdx.x, lane = t & 31, wave = t >> 5;
  const int qb = blockIdx.x, h = blockIdx.y;
  const int q0 = qb * 128 + wave * 16;
  const int nn = lane & 15, half = lane >> 4;

  // Q fragments (16 rows x 64 k = 2 A-frags), loaded once. Q pre-scaled 1/8.
  v16bf aq[2];
  {
    const bf16_t* qrow = Qb + (size_t)(q0 + nn) * DMODEL + h * DHEAD;
    #pragma unroll
    for (int kk = 0; kk < 2; ++kk) {
      v8bf lo = *(const v8bf*)(qrow + kk * 32 + half * 8);
      v8bf hi = *(const v8bf*)(qrow + kk * 32 + 16 + half * 8);
      aq[kk] = pack16(lo, hi);
    }
  }

  float mrow[8], lrow[8];
  v8f acc[4];
  #pragma unroll
  for (int r = 0; r < 8; ++r) { mrow[r] = -1e30f; lrow[r] = 0.0f; }
  #pragma unroll
  for (int j = 0; j < 4; ++j) acc[j] = (v8f)(0.0f);

  auto stageKV = [&](int buf, int kb) {
    int r = t >> 3, o = (t & 7) * 8;           // K: 32 x 64 -> 256 chunks
    cp_async16(&Ks[buf][r][o], &Kb[(size_t)(kb + r) * DMODEL + h * DHEAD + o]);
    int r2 = t >> 2, o2 = (t & 3) * 8;         // Vt: 64 x 32 -> 256 chunks
    cp_async16(&Vts[buf][r2][o2], &Vtb[(size_t)(h * DHEAD + r2) * SEQ + kb + o2]);
  };

  stageKV(0, 0);
  const int NIT = SEQ / 32;
  for (int it = 0; it < NIT; ++it) {
    const int cur = it & 1;
    wait_async_all();
    __syncthreads();
    if (it + 1 < NIT) stageKV(cur ^ 1, (it + 1) * 32);

    // S = Q * K^T for 32 keys: two 16-col tiles, each 2 k-steps of 32
    v8f s0 = (v8f)(0.0f), s1 = (v8f)(0.0f);
    s0 = wmma_bf16(aq[0], load_bt_frag(&Ks[cur][0][0],  72, lane, 0),  s0);
    s0 = wmma_bf16(aq[1], load_bt_frag(&Ks[cur][0][0],  72, lane, 32), s0);
    s1 = wmma_bf16(aq[0], load_bt_frag(&Ks[cur][16][0], 72, lane, 0),  s1);
    s1 = wmma_bf16(aq[1], load_bt_frag(&Ks[cur][16][0], 72, lane, 32), s1);

    // online softmax; row (r + 8*half)'s 16 cols live across this half's lanes
    float p0[8], p1[8];
    #pragma unroll
    for (int r = 0; r < 8; ++r) {
      float mx = redmax16(fmaxf(s0[r], s1[r]));
      float mn = fmaxf(mrow[r], mx);
      float alpha = __expf(mrow[r] - mn);
      mrow[r] = mn;
      p0[r] = __expf(s0[r] - mn);
      p1[r] = __expf(s1[r] - mn);
      lrow[r] = lrow[r] * alpha + redsum16(p0[r] + p1[r]);
      #pragma unroll
      for (int j = 0; j < 4; ++j) acc[j][r] *= alpha;
    }

    // C-layout P -> A-layout via per-wave LDS bounce
    bf16_t* P = &Plds[wave][0][0];
    #pragma unroll
    for (int r = 0; r < 8; ++r) {
      const int row = r + 8 * half;
      P[row * 40 + nn]      = (bf16_t)p0[r];
      P[row * 40 + 16 + nn] = (bf16_t)p1[r];
    }
    asm volatile("s_wait_dscnt 0" ::: "memory");  // wave-local bounce visible

    v16bf ap;
    {
      const bf16_t* prow = &Plds[wave][nn][0];
      v8bf lo = *(const v8bf*)(prow + half * 8);
      v8bf hi = *(const v8bf*)(prow + 16 + half * 8);
      ap = pack16(lo, hi);
    }

    // O += P * V : 4 d-tiles of 16, k = 32 keys
    #pragma unroll
    for (int j = 0; j < 4; ++j) {
      v16bf bv = load_bt_frag(&Vts[cur][j * 16][0], 40, lane, 0);
      acc[j] = wmma_bf16(ap, bv, acc[j]);
    }
  }

  // normalize and store merged-head bf16 output [s][1024]
  #pragma unroll
  for (int r = 0; r < 8; ++r) {
    const float inv = 1.0f / lrow[r];
    const int row = q0 + r + 8 * half;
    #pragma unroll
    for (int j = 0; j < 4; ++j)
      attnb[(size_t)row * DMODEL + h * DHEAD + j * 16 + nn] = (bf16_t)(acc[j][r] * inv);
  }
}

// ------------------------------------------------------------------ launch ---
extern "C" void kernel_launch(void* const* d_in, const int* in_sizes, int n_in,
                              void* d_out, int out_size, void* d_ws, size_t ws_size,
                              hipStream_t stream) {
  const float* x   = (const float*)d_in[0];
  const float* W_Q = (const float*)d_in[1];
  const float* W_K = (const float*)d_in[2];
  const float* W_V = (const float*)d_in[3];
  const float* W_O = (const float*)d_in[4];

  bf16_t* p = (bf16_t*)d_ws;                 // 48 MiB total (fits 192 MB L2)
  bf16_t* xb    = p; p += (size_t)SEQ * DMODEL;
  bf16_t* wqb   = p; p += (size_t)DMODEL * DMODEL;
  bf16_t* wkb   = p; p += (size_t)DMODEL * DMODEL;
  bf16_t* wvb   = p; p += (size_t)DMODEL * DMODEL;
  bf16_t* wob   = p; p += (size_t)DMODEL * DMODEL;
  bf16_t* Qb    = p; p += (size_t)SEQ * DMODEL;
  bf16_t* Kbuf  = p; p += (size_t)SEQ * DMODEL;
  bf16_t* Vtb   = p; p += (size_t)DMODEL * SEQ;
  bf16_t* attnb = p; p += (size_t)SEQ * DMODEL;

  const int nx = SEQ * DMODEL, nw = DMODEL * DMODEL;
  cvt_f32_bf16<<<4096, 256, 0, stream>>>(x,   xb,  nx);
  cvt_f32_bf16<<<1024, 256, 0, stream>>>(W_Q, wqb, nw);
  cvt_f32_bf16<<<1024, 256, 0, stream>>>(W_K, wkb, nw);
  cvt_f32_bf16<<<1024, 256, 0, stream>>>(W_V, wvb, nw);
  cvt_f32_bf16<<<1024, 256, 0, stream>>>(W_O, wob, nw);

  dim3 ggrid(DMODEL / 64, SEQ / 256);        // (16, 16)
  gemm_bf16<<<ggrid, 256, 0, stream>>>(xb, wqb, Qb,   nullptr, 0);
  gemm_bf16<<<ggrid, 256, 0, stream>>>(xb, wkb, Kbuf, nullptr, 1);
  gemm_bf16<<<ggrid, 256, 0, stream>>>(xb, wvb, Vtb,  nullptr, 2);

  attn_flash<<<dim3(SEQ / 128, NHEAD), 256, 0, stream>>>(Qb, Kbuf, Vtb, attnb);

  gemm_bf16<<<ggrid, 256, 0, stream>>>(attnb, wob, nullptr, (float*)d_out, 3);
}
